// WaterMDNetNew_14499809591857
// MI455X (gfx1250) — matrix operand order repaired and checked by reading, not compiled, not run on hardware
//
#include <hip/hip_runtime.h>

typedef _Float16 h16;
typedef __attribute__((ext_vector_type(16))) _Float16 v16h;
typedef __attribute__((ext_vector_type(8)))  float    v8f;

#define NNODES 50000
#define NEDGES 800000
#define DMODEL 128
#define MATSZ  16384            // 128*128 elements per weight matrix
#define ETILES (NEDGES / 16)    // 50000
#define NTILES (NNODES / 16)    // 3125

// ---------------------------------------------------------------------------
// WMMA helpers
// ---------------------------------------------------------------------------
__device__ inline v8f wmma_f16(v16h a, v16h b, v8f c) {
  return __builtin_amdgcn_wmma_f32_16x16x32_f16(false, a, false, b, (short)0, c,
                                                false, false);
}

// A-fragment (16x32 f16) from an LDS [16][128] f16 tile, K-chunk kc (K=32*kc..)
// ISA layout: lanes 0-15 row M=lane, halves 0-7 = K kb..kb+7, 8-15 = K kb+16..kb+23
//             lanes 16-31 same rows, kb += 8
__device__ inline v16h load_a_frag(const h16* sc, int lane, int kc) {
  int row = lane & 15;
  int hi  = lane >> 4;
  const uint4* u = (const uint4*)(sc + row * DMODEL);
  int q = kc * 4 + hi;              // uint4 index: covers halves K=kb..kb+7
  union { v16h v; uint4 u[2]; } r;
  r.u[0] = u[q];
  r.u[1] = u[q + 2];                // halves K=kb+16..kb+23
  return r.v;
}

// B-fragment from pre-swizzled weight storage: frag (nt,kc,lane) is 16
// contiguous halves (32 bytes) -> two ds_load_b128.
__device__ inline v16h load_b_frag(const h16* w, int lane, int nt, int kc) {
  return *((const v16h*)w + ((nt * 4 + kc) * 32 + lane));
}

__device__ inline float silu(float x) { return x / (1.f + __expf(-x)); }

// C layout: lane l holds (M = r + 8*(l>>4), N = nt*16 + (l&15)) for r=0..7
__device__ inline void store_c_silu(h16* out, int lane, int nt, v8f acc) {
  int n  = nt * 16 + (lane & 15);
  int mb = (lane >> 4) * 8;
#pragma unroll
  for (int r = 0; r < 8; ++r)
    out[(mb + r) * DMODEL + n] = (h16)silu(acc[r]);
}

// Full [16,128] GEMM stage: acc[8] += sum_s A_s @ B_s   (K = 128)
// Software-pipelined: B fragment for nt+1 (and the next chain's A / first B)
// is issued before the WMMA consuming the current fragment, so ds_load
// latency hides under the matrix pipe instead of s_wait_dscnt 0 stalls.
template <int NS>
__device__ inline void gemm_stage(int lane, v8f acc[8],
                                  const h16* a0, const h16* w0,
                                  const h16* a1, const h16* w1,
                                  const h16* a2, const h16* w2) {
  const h16* As[3] = {a0, a1, a2};
  const h16* Wm[3] = {w0, w1, w2};

  v16h af = load_a_frag(As[0], lane, 0);
  v16h bc = load_b_frag(Wm[0], lane, 0, 0);
#pragma unroll
  for (int it = 0; it < 4 * NS; ++it) {
    const int s  = it % NS,        kc  = it / NS;
    const int sn = (it + 1) % NS,  kcn = (it + 1) / NS;
    v16h af_n = af, bc_n = bc;
#pragma unroll
    for (int nt = 0; nt < 8; ++nt) {
      v16h bn = bc;
      if (nt < 7) {
        bn = load_b_frag(Wm[s], lane, nt + 1, kc);
      } else if (it < 4 * NS - 1) {
        af_n = load_a_frag(As[sn], lane, kcn);
        bc_n = load_b_frag(Wm[sn], lane, 0, kcn);
      }
      acc[nt] = wmma_f16(af, bc, acc[nt]);
      bc = bn;
    }
    af = af_n;
    bc = bc_n;
  }
}

// ---------------------------------------------------------------------------
// Weight packing: f32 [K=128][N=128] row-major -> f16 B-fragment order
// dst[((nt*4+kc)*32 + lane)*16 + j] = W[kc*32 + (lane>>4)*8 + (j&7) + (j>>3)*16][nt*16 + (lane&15)]
// ---------------------------------------------------------------------------
__global__ void pack_w_kernel(const float* __restrict__ src, h16* __restrict__ dst) {
  int i = blockIdx.x * blockDim.x + threadIdx.x;
  if (i >= MATSZ) return;
  int j    = i & 15;
  int lane = (i >> 4) & 31;
  int fr   = i >> 9;            // nt*4 + kc
  int kc   = fr & 3;
  int nt   = fr >> 2;
  int n = nt * 16 + (lane & 15);
  int k = kc * 32 + (lane >> 4) * 8 + (j & 7) + (j >> 3) * 16;
  dst[i] = (h16)src[k * DMODEL + n];
}

// ---------------------------------------------------------------------------
// Encoder: h0 = x @ enc_W + enc_b    [N,16] x [16,128]
// ---------------------------------------------------------------------------
__global__ void enc_kernel(const float* __restrict__ x, const float* __restrict__ W,
                           const float* __restrict__ b, float* __restrict__ h) {
  int i = blockIdx.x * blockDim.x + threadIdx.x;   // exactly N*128 threads
  int n = i >> 7, d = i & 127;
  const float* xr = x + n * 16;
  float s = b[d];
#pragma unroll
  for (int k = 0; k < 16; ++k) s = fmaf(xr[k], W[k * DMODEL + d], s);
  h[i] = s;
}

// ---------------------------------------------------------------------------
// LayerNorm: hn(f16) = (h-mu)*rsqrt(var+eps)*g + b   one row per wave
// ---------------------------------------------------------------------------
__global__ void ln_kernel(const float* __restrict__ h, const float* __restrict__ g,
                          const float* __restrict__ b, h16* __restrict__ hn) {
  int row  = (blockIdx.x * blockDim.x + threadIdx.x) >> 5;
  if (row >= NNODES) return;
  int lane = threadIdx.x & 31;
  float4 v = ((const float4*)(h + (size_t)row * DMODEL))[lane];
  float s = v.x + v.y + v.z + v.w;
  float q = v.x * v.x + v.y * v.y + v.z * v.z + v.w * v.w;
#pragma unroll
  for (int m = 16; m >= 1; m >>= 1) {
    s += __shfl_xor(s, m, 32);
    q += __shfl_xor(q, m, 32);
  }
  float mu = s * (1.f / 128.f);
  float rs = rsqrtf(q * (1.f / 128.f) - mu * mu + 1e-5f);
  float4 gg = ((const float4*)g)[lane];
  float4 bb = ((const float4*)b)[lane];
  union { h16 o[4]; uint2 u; } pk;
  pk.o[0] = (h16)((v.x - mu) * rs * gg.x + bb.x);
  pk.o[1] = (h16)((v.y - mu) * rs * gg.y + bb.y);
  pk.o[2] = (h16)((v.z - mu) * rs * gg.z + bb.z);
  pk.o[3] = (h16)((v.w - mu) * rs * gg.w + bb.w);
  ((uint2*)(hn + (size_t)row * DMODEL))[lane] = pk.u;
}

// ---------------------------------------------------------------------------
// Edge kernel: per 16-edge tile, 4 chained f16 WMMA GEMMs + atomic segment-sum
// LDS: 6 weight mats (192KB, block-shared) + 3x4KB scratch per wave + indices
// ---------------------------------------------------------------------------
#define EDGE_WAVES 8
#define EDGE_SMEM  (6 * MATSZ * 2 + EDGE_WAVES * 3 * 2048 * 2 + EDGE_WAVES * 32 * 4)

__global__ __launch_bounds__(EDGE_WAVES * 32, 1)
void edge_kernel(const float* __restrict__ e, const h16* __restrict__ hn,
                 float* __restrict__ agg,
                 const int* __restrict__ srcI, const int* __restrict__ dstI,
                 const h16* __restrict__ wpackL,
                 const float* __restrict__ eb1, const float* __restrict__ eb2,
                 const float* __restrict__ sb,  const float* __restrict__ db,
                 const float* __restrict__ tb1, const float* __restrict__ tb2) {
  extern __shared__ __align__(32) char smem[];
  h16* wlds = (h16*)smem;                           // 6*16384 halves
  {
    const uint4* s = (const uint4*)wpackL;
    uint4* d = (uint4*)wlds;
    for (int i = threadIdx.x; i < 6 * 2048; i += blockDim.x) d[i] = s[i];
  }
  __syncthreads();

  int wave = threadIdx.x >> 5, lane = threadIdx.x & 31;
  h16* scr = wlds + 6 * MATSZ + wave * 3 * 2048;
  h16* B0 = scr;                 // e tile -> u1 -> t
  h16* B1 = scr + 2048;          // hn[src]   (kept for message)
  h16* B2 = scr + 4096;          // hn[dst] -> u3
  int* sI = (int*)(wlds + 6 * MATSZ + EDGE_WAVES * 3 * 2048) + wave * 32;

  const h16 *W1 = wlds,             *W2 = wlds + MATSZ,
            *Ws = wlds + 2 * MATSZ, *Wd = wlds + 3 * MATSZ,
            *Wt1 = wlds + 4 * MATSZ, *Wt2 = wlds + 5 * MATSZ;

  int nl = lane & 15, mb = (lane >> 4) * 8;
  int gw  = (blockIdx.x * blockDim.x + threadIdx.x) >> 5;
  int gws = (gridDim.x * blockDim.x) >> 5;

  for (int tile = gw; tile < ETILES; tile += gws) {
    int ebase = tile * 16;
    if (lane < 16) {
      sI[lane]      = srcI[ebase + lane];
      sI[16 + lane] = dstI[ebase + lane];
    }
    // stage e rows (f32 -> f16) into B0: tile rows are contiguous
    const float4* e4 = (const float4*)(e + (size_t)ebase * DMODEL);
    for (int t = lane; t < 512; t += 32) {
      float4 v = e4[t];
      union { h16 o[4]; uint2 u; } pk;
      pk.o[0] = (h16)v.x; pk.o[1] = (h16)v.y;
      pk.o[2] = (h16)v.z; pk.o[3] = (h16)v.w;
      ((uint2*)B0)[t] = pk.u;
    }
    // gather hn[src] -> B1, hn[dst] -> B2 (f16, 256B per row)
#pragma unroll 4
    for (int m = 0; m < 16; ++m) {
      int si = sI[m], di = sI[16 + m];
      ((uint2*)(B1 + m * DMODEL))[lane] = ((const uint2*)(hn + (size_t)si * DMODEL))[lane];
      ((uint2*)(B2 + m * DMODEL))[lane] = ((const uint2*)(hn + (size_t)di * DMODEL))[lane];
    }

    v8f acc[8];
    // GEMM1: u1 = silu(e @ eW1 + eb1)  -> B0
#pragma unroll
    for (int nt = 0; nt < 8; ++nt) {
      float bv = eb1[nt * 16 + nl];
#pragma unroll
      for (int r = 0; r < 8; ++r) acc[nt][r] = bv;
    }
    gemm_stage<1>(lane, acc, B0, W1, nullptr, nullptr, nullptr, nullptr);
#pragma unroll
    for (int nt = 0; nt < 8; ++nt) store_c_silu(B0, lane, nt, acc[nt]);

    // GEMM2: t = silu(u1@eW2 + hs@sW + hd@dW + eb2+sb+db)  -> B0
#pragma unroll
    for (int nt = 0; nt < 8; ++nt) {
      int n = nt * 16 + nl;
      float bv = eb2[n] + sb[n] + db[n];
#pragma unroll
      for (int r = 0; r < 8; ++r) acc[nt][r] = bv;
    }
    gemm_stage<3>(lane, acc, B0, W2, B1, Ws, B2, Wd);
#pragma unroll
    for (int nt = 0; nt < 8; ++nt) store_c_silu(B0, lane, nt, acc[nt]);

    // GEMM3: u3 = silu(t @ tW1 + tb1)  -> B2
#pragma unroll
    for (int nt = 0; nt < 8; ++nt) {
      float bv = tb1[nt * 16 + nl];
#pragma unroll
      for (int r = 0; r < 8; ++r) acc[nt][r] = bv;
    }
    gemm_stage<1>(lane, acc, B0, Wt1, nullptr, nullptr, nullptr, nullptr);
#pragma unroll
    for (int nt = 0; nt < 8; ++nt) store_c_silu(B2, lane, nt, acc[nt]);

    // GEMM4: e_emb = u3 @ tW2 + tb2 ; msg = h_src * e_emb ; atomic scatter
#pragma unroll
    for (int nt = 0; nt < 8; ++nt) {
      float bv = tb2[nt * 16 + nl];
#pragma unroll
      for (int r = 0; r < 8; ++r) acc[nt][r] = bv;
    }
    gemm_stage<1>(lane, acc, B2, Wt2, nullptr, nullptr, nullptr, nullptr);
#pragma unroll
    for (int nt = 0; nt < 8; ++nt) {
      int n = nt * 16 + nl;
#pragma unroll
      for (int r = 0; r < 8; ++r) {
        int M = mb + r;
        float msg = (float)B1[M * DMODEL + n] * acc[nt][r];
        int dn = sI[16 + M];
        __hip_atomic_fetch_add(&agg[(size_t)dn * DMODEL + n], msg,
                               __ATOMIC_RELAXED, __HIP_MEMORY_SCOPE_AGENT);
      }
    }
  }
}

// ---------------------------------------------------------------------------
// Node kernel: h += silu(hn@pdW + agg@peW + pdb+peb) @ phW + phb
// ---------------------------------------------------------------------------
#define NODE_WAVES 8
#define NODE_SMEM  (3 * MATSZ * 2 + NODE_WAVES * 2 * 2048 * 2)

__global__ __launch_bounds__(NODE_WAVES * 32, 1)
void node_kernel(const h16* __restrict__ hn, const float* __restrict__ agg,
                 float* __restrict__ h, const h16* __restrict__ wpackN,
                 const float* __restrict__ pdb, const float* __restrict__ peb,
                 const float* __restrict__ phb) {
  extern __shared__ __align__(32) char smem[];
  h16* wlds = (h16*)smem;                          // 3*16384 halves
  {
    const uint4* s = (const uint4*)wpackN;
    uint4* d = (uint4*)wlds;
    for (int i = threadIdx.x; i < 3 * 2048; i += blockDim.x) d[i] = s[i];
  }
  __syncthreads();

  int wave = threadIdx.x >> 5, lane = threadIdx.x & 31;
  h16* scr = wlds + 3 * MATSZ + wave * 2 * 2048;
  h16* B0 = scr;                 // hn tile -> u
  h16* B1 = scr + 2048;          // agg tile (f16)
  const h16 *Wpd = wlds, *Wpe = wlds + MATSZ, *Wph = wlds + 2 * MATSZ;

  int nl = lane & 15, mb = (lane >> 4) * 8;
  int gw  = (blockIdx.x * blockDim.x + threadIdx.x) >> 5;
  int gws = (gridDim.x * blockDim.x) >> 5;

  for (int tile = gw; tile < NTILES; tile += gws) {
    int base = tile * 16;
    const uint4* s = (const uint4*)(hn + (size_t)base * DMODEL);
    for (int t = lane; t < 256; t += 32) ((uint4*)B0)[t] = s[t];
    const float4* a4 = (const float4*)(agg + (size_t)base * DMODEL);
    for (int t = lane; t < 512; t += 32) {
      float4 v = a4[t];
      union { h16 o[4]; uint2 u; } pk;
      pk.o[0] = (h16)v.x; pk.o[1] = (h16)v.y;
      pk.o[2] = (h16)v.z; pk.o[3] = (h16)v.w;
      ((uint2*)B1)[t] = pk.u;
    }

    v8f acc[8];
    // u = silu(hn@pdW + agg@peW + pdb + peb) -> B0
#pragma unroll
    for (int nt = 0; nt < 8; ++nt) {
      int n = nt * 16 + nl;
      float bv = pdb[n] + peb[n];
#pragma unroll
      for (int r = 0; r < 8; ++r) acc[nt][r] = bv;
    }
    gemm_stage<2>(lane, acc, B0, Wpd, B1, Wpe, nullptr, nullptr);
#pragma unroll
    for (int nt = 0; nt < 8; ++nt) store_c_silu(B0, lane, nt, acc[nt]);

    // out = u @ phW + phb ; h += out  (residual, exclusive tile -> plain RMW)
#pragma unroll
    for (int nt = 0; nt < 8; ++nt) {
      float bv = phb[nt * 16 + nl];
#pragma unroll
      for (int r = 0; r < 8; ++r) acc[nt][r] = bv;
    }
    gemm_stage<1>(lane, acc, B0, Wph, nullptr, nullptr, nullptr, nullptr);
#pragma unroll
    for (int nt = 0; nt < 8; ++nt) {
      int n = nt * 16 + nl;
#pragma unroll
      for (int r = 0; r < 8; ++r) {
        size_t off = (size_t)(base + mb + r) * DMODEL + n;
        h[off] = h[off] + acc[nt][r];
      }
    }
  }
}

// ---------------------------------------------------------------------------
// Launcher
// ---------------------------------------------------------------------------
extern "C" void kernel_launch(void* const* d_in, const int* in_sizes, int n_in,
                              void* d_out, int out_size, void* d_ws, size_t ws_size,
                              hipStream_t stream) {
  (void)in_sizes; (void)n_in; (void)out_size; (void)ws_size;
  const float* x      = (const float*)d_in[0];
  const float* e      = (const float*)d_in[1];
  const float* enc_W  = (const float*)d_in[2];
  const float* enc_b  = (const float*)d_in[3];
  const float* ln_g   = (const float*)d_in[4];
  const float* ln_b   = (const float*)d_in[5];
  const float* ea_W1  = (const float*)d_in[6];
  const float* ea_b1  = (const float*)d_in[7];
  const float* ea_W2  = (const float*)d_in[8];
  const float* ea_b2  = (const float*)d_in[9];
  const float* sa_W   = (const float*)d_in[10];
  const float* sa_b   = (const float*)d_in[11];
  const float* da_W   = (const float*)d_in[12];
  const float* da_b   = (const float*)d_in[13];
  const float* th_W1  = (const float*)d_in[14];
  const float* th_b1  = (const float*)d_in[15];
  const float* th_W2  = (const float*)d_in[16];
  const float* th_b2  = (const float*)d_in[17];
  const float* pd_W   = (const float*)d_in[18];
  const float* pd_b   = (const float*)d_in[19];
  const float* pe_W   = (const float*)d_in[20];
  const float* pe_b   = (const float*)d_in[21];
  const float* ph_W   = (const float*)d_in[22];
  const float* ph_b   = (const float*)d_in[23];
  const int* src_idx  = (const int*)d_in[24];
  const int* dst_idx  = (const int*)d_in[25];

  // workspace layout
  h16*   wpack = (h16*)d_ws;                        // 18 * 16384 halves
  h16*   hn    = wpack + 18 * MATSZ;                // N*128 halves
  float* agg   = (float*)(hn + (size_t)NNODES * DMODEL);  // N*128 floats
  float* h     = (float*)d_out;

  // 1) pack all weights to f16 fragment order
  const float* Wsrc[9] = {ea_W1, ea_W2, sa_W, da_W, th_W1, th_W2, pd_W, pe_W, ph_W};
  for (int l = 0; l < 2; ++l)
    for (int m = 0; m < 9; ++m)
      pack_w_kernel<<<MATSZ / 256, 256, 0, stream>>>(
          Wsrc[m] + (size_t)l * MATSZ, wpack + ((size_t)l * 9 + m) * MATSZ);

  // 2) encoder
  enc_kernel<<<(NNODES * DMODEL) / 256, 256, 0, stream>>>(x, enc_W, enc_b, h);

  // 3) layers
  for (int l = 0; l < 2; ++l) {
    ln_kernel<<<(NNODES * 32 + 255) / 256, 256, 0, stream>>>(
        h, ln_g + l * DMODEL, ln_b + l * DMODEL, hn);
    (void)hipMemsetAsync(agg, 0, (size_t)NNODES * DMODEL * sizeof(float), stream);
    edge_kernel<<<400, EDGE_WAVES * 32, EDGE_SMEM, stream>>>(
        e, hn, agg, src_idx, dst_idx, wpack + (size_t)l * 9 * MATSZ,
        ea_b1 + l * DMODEL, ea_b2 + l * DMODEL, sa_b + l * DMODEL,
        da_b + l * DMODEL, th_b1 + l * DMODEL, th_b2 + l * DMODEL);
    node_kernel<<<256, NODE_WAVES * 32, NODE_SMEM, stream>>>(
        hn, agg, h, wpack + (size_t)(l * 9 + 6) * MATSZ,
        pd_b + l * DMODEL, pe_b + l * DMODEL, ph_b + l * DMODEL);
  }
}
